// NonRigidMotionMLP_63471026700447
// MI455X (gfx1250) — matrix operand order repaired and checked by reading, not compiled, not run on hardware
//
#include <hip/hip_runtime.h>
#include <hip/hip_bf16.h>

typedef __attribute__((ext_vector_type(16))) _Float16 v16h;
typedef __attribute__((ext_vector_type(8)))  _Float16 v8h;
typedef __attribute__((ext_vector_type(8)))  float    v8f;

#define N_PTS   524288
#define NCOND   69
#define K0      96      // 85 inputs zero-padded to 96 (3 K-chunks of 32)
#define WIDTH   128
#define MTILE   64
#define NBLOBS  188     // 24 (L0) + 5*32 (L1..5) + 4 (L6)
#define HASH_MASK 0x7FFFFu   // all hashed levels have size 2^19

__constant__ int      c_res[8]   = {16, 32, 64, 128, 256, 512, 1024, 2048};
__constant__ unsigned c_size[8]  = {4920u, 35944u, 274632u, 524288u, 524288u, 524288u, 524288u, 524288u};
__constant__ unsigned c_off[8]   = {0u, 4920u, 40864u, 315496u, 839784u, 1364072u, 1888360u, 2412648u};

// ---------------------------------------------------------------------------
// Kernel 1: repack fp32 weights into f16 WMMA *A*-fragment blobs (W^T tiles).
// We compute D = W^T x act^T, so the weight tile is the A operand:
// 16x32 (rows = out-feature n, cols = k). In the contiguous k order used here,
// lane l holds row n = l%16; elements 0..15 = K {2*koff..2*koff+15},
// koff = (l>=16 ? 8 : 0). 1 KB per blob.
// ---------------------------------------------------------------------------
__global__ void prep_weights(const float* __restrict__ W0, const float* __restrict__ W1,
                             const float* __restrict__ W2, const float* __restrict__ W3,
                             const float* __restrict__ W4, const float* __restrict__ W5,
                             const float* __restrict__ W6, _Float16* __restrict__ dst) {
    int b = blockIdx.x;          // blob id 0..187
    int l = threadIdx.x;         // lane 0..31
    const float* W;
    int Kact, Nact, kc, nt;
    if (b < 24) {                                  // layer 0: W0 [85 x 128]
        W = W0; Kact = 85; Nact = 128; kc = b >> 3; nt = b & 7;
    } else if (b < 184) {                          // layers 1..5: [128 x 128]
        int r = b - 24; int ly = r >> 5; int r2 = r & 31;
        W = (ly == 0) ? W1 : (ly == 1) ? W2 : (ly == 2) ? W3 : (ly == 3) ? W4 : W5;
        Kact = 128; Nact = 128; kc = r2 >> 3; nt = r2 & 7;
    } else {                                       // layer 6: W6 [128 x 3] padded to N=16
        W = W6; Kact = 128; Nact = 3; kc = b - 184; nt = 0;
    }
    const int n    = nt * 16 + (l & 15);
    const int koff = (l >= 16) ? 8 : 0;
    _Float16* out = dst + (size_t)b * 512 + (size_t)l * 16;
#pragma unroll
    for (int e = 0; e < 16; ++e) {
        const int k = kc * 32 + 2 * koff + e;
        float v = (k < Kact && n < Nact) ? W[(size_t)k * Nact + n] : 0.0f;
        out[e] = (_Float16)v;
    }
}

// ---------------------------------------------------------------------------
// One MLP hidden layer for one wave, transposed product:
//   D(16n x 16m) = W^T(16n x 32k) * act^T(32k x 16m)
// A = weight blob (registers, preloaded for the whole layer).
// B = activation fragment from LDS: lane l -> point row m = mt*16 + l%16,
//     16 contiguous K halves at koff2 = (l>=16 ? 16 : 0)  (one 32B read).
// D store: lane l owns point row m and 8 consecutive feature cols
//     wv*16 + mbase + {0..7} -> one packed ds_store_b128 per row-tile.
// ---------------------------------------------------------------------------
template <int KCN>
__device__ __forceinline__ void mlp_layer(const _Float16* __restrict__ wts, int blobBase,
                                          int wv, int lane, int koff2, int mbase,
                                          const _Float16 (*__restrict__ actIn)[WIDTH],
                                          _Float16 (*__restrict__ actOut)[WIDTH],
                                          const float* __restrict__ bp) {
    v16h aw[KCN];
#pragma unroll
    for (int kc = 0; kc < KCN; ++kc)
        aw[kc] = *(const v16h*)(wts + (size_t)(blobBase + kc * 8 + wv) * 512 + (size_t)lane * 16);

    v8f acc[4] = {};
#pragma unroll
    for (int kc = 0; kc < KCN; ++kc) {
        union { v16h v; v8h h[2]; } bf[4];
#pragma unroll
        for (int mt = 0; mt < 4; ++mt) {
            const _Float16* p = &actIn[mt * 16 + (lane & 15)][kc * 32 + koff2];
            bf[mt].h[0] = *(const v8h*)p;
            bf[mt].h[1] = *(const v8h*)(p + 8);
        }
#pragma unroll
        for (int mt = 0; mt < 4; ++mt)
            acc[mt] = __builtin_amdgcn_wmma_f32_16x16x32_f16(
                false, aw[kc], false, bf[mt].v, (short)0, acc[mt], false, false);
    }

    // bias per feature (the D-tile M dimension): 8 consecutive features
    const float4 bA = *(const float4*)&bp[wv * 16 + mbase];
    const float4 bB = *(const float4*)&bp[wv * 16 + mbase + 4];
    const float bias[8] = {bA.x, bA.y, bA.z, bA.w, bB.x, bB.y, bB.z, bB.w};
#pragma unroll
    for (int mt = 0; mt < 4; ++mt) {
        v8h hs;
#pragma unroll
        for (int v = 0; v < 8; ++v) {
            float val = acc[mt][v] + bias[v];
            val = val > 0.0f ? val : 0.0f;                    // ReLU
            hs[v] = (_Float16)val;
        }
        *(v8h*)&actOut[mt * 16 + (lane & 15)][wv * 16 + mbase] = hs;   // one b128 store
    }
}

// ---------------------------------------------------------------------------
// Kernel 2: fused grid-encode + 7-layer MLP. 256 threads = 8 waves, 64 points.
// ---------------------------------------------------------------------------
__launch_bounds__(256)
__global__ void fused_encode_mlp(const float* __restrict__ xyz,
                                 const float* __restrict__ cond,
                                 const float* __restrict__ emb,
                                 const _Float16* __restrict__ wts,
                                 const float* __restrict__ b0, const float* __restrict__ b1,
                                 const float* __restrict__ b2, const float* __restrict__ b3,
                                 const float* __restrict__ b4, const float* __restrict__ b5,
                                 const float* __restrict__ b6,
                                 float* __restrict__ out) {
    __shared__ __align__(16) _Float16 act[2][MTILE][WIDTH];   // 32 KB double buffer

    const int tid     = threadIdx.x;
    const int lane    = tid & 31;
    const int wv      = tid >> 5;           // wave 0..7
    const int rowBase = blockIdx.x * MTILE;

    // ---- Phase 1a: condition_code -> act[0] cols 0..68 (f16), b128 loads ----
    {
        const float4* src = (const float4*)(cond + (size_t)rowBase * NCOND); // 16B aligned (64*69*4)
        for (int i = tid; i < (MTILE * NCOND) / 4; i += 256) {               // 1104 float4s
            const float4 v = src[i];
            const int f = i * 4;
#pragma unroll
            for (int e = 0; e < 4; ++e) {
                const int fe = f + e;
                const int r = fe / NCOND;              // const divisor -> mulhi
                const int c = fe - r * NCOND;
                const float val = (e == 0) ? v.x : (e == 1) ? v.y : (e == 2) ? v.z : v.w;
                act[0][r][c] = (_Float16)val;
            }
        }
    }
    // zero-pad cols 85..95
    for (int i = tid; i < MTILE * (K0 - 85); i += 256) {
        int r = i / (K0 - 85), c = 85 + (i - r * (K0 - 85));
        act[0][r][c] = (_Float16)0.0f;
    }

    // ---- Phase 1b: hash-grid encode -> act[0] cols 69..84 ----
    {
        const int p   = tid >> 2;        // point 0..63
        const int sub = tid & 3;         // 2 levels per thread
        const size_t gp = (size_t)(rowBase + p);
        const float x = (xyz[gp * 3 + 0] + 1.0f) * 0.5f;
        const float y = (xyz[gp * 3 + 1] + 1.0f) * 0.5f;
        const float z = (xyz[gp * 3 + 2] + 1.0f) * 0.5f;
#pragma unroll
        for (int j = 0; j < 2; ++j) {
            const int lev = sub * 2 + j;
            const float resf = (float)c_res[lev];
            const float px = x * resf + 0.5f, py = y * resf + 0.5f, pz = z * resf + 0.5f;
            const unsigned ix = (unsigned)px, iy = (unsigned)py, iz = (unsigned)pz; // px>=0: trunc==floor
            const float tx = px - (float)ix, ty = py - (float)iy, tz = pz - (float)iz;
            const unsigned size = c_size[lev];
            const unsigned off  = c_off[lev];
            const unsigned r1   = (unsigned)c_res[lev] + 1u;
            const bool dense    = (lev < 3);
            float f0 = 0.0f, f1 = 0.0f;
#pragma unroll
            for (int c = 0; c < 8; ++c) {
                const unsigned bx = (unsigned)(c & 1), by = (unsigned)((c >> 1) & 1), bz = (unsigned)((c >> 2) & 1);
                const unsigned cx = ix + bx, cy = iy + by, cz = iz + bz;
                unsigned idx;
                if (dense) idx = (cx + cy * r1 + cz * r1 * r1) % size;     // small dense tables
                else       idx = (cx ^ (cy * 2654435761u) ^ (cz * 805459861u)) & HASH_MASK; // 2^19
                const float w = (bx ? tx : 1.0f - tx) * (by ? ty : 1.0f - ty) * (bz ? tz : 1.0f - tz);
                const float* e = emb + 2u * (off + idx);
                f0 += w * e[0];
                f1 += w * e[1];
            }
            act[0][p][NCOND + lev * 2 + 0] = (_Float16)f0;
            act[0][p][NCOND + lev * 2 + 1] = (_Float16)f1;
        }
    }
    __syncthreads();

    // ---- Phase 2: hidden layers (WMMA f16 -> f32 acc, transposed product) ----
    const int koff2 = (lane >= 16) ? 16 : 0;  // B-fragment K offset (contiguous k order, as in A blobs)
    const int mbase = (lane >= 16) ? 8 : 0;   // D-tile M (feature) offset per ISA layout

    // layer 0: K = 96 (3 chunks)
    mlp_layer<3>(wts, 0, wv, lane, koff2, mbase, act[0], act[1], b0);
    __syncthreads();

    // layers 1..5: K = 128 (4 chunks), double-buffered
    int buf = 1;
    for (int l = 1; l < 6; ++l) {
        const float* bp = (l == 1) ? b1 : (l == 2) ? b2 : (l == 3) ? b3 : (l == 4) ? b4 : b5;
        mlp_layer<4>(wts, 24 + (l - 1) * 32, wv, lane, koff2, mbase,
                     act[buf], act[buf ^ 1], bp);
        buf ^= 1;
        __syncthreads();
    }

    // ---- Phase 3: final layer 128 -> 3 (padded N=16), waves 0..3 ----
    if (wv < 4) {
        const int mt = wv;
        v16h aw[4];
#pragma unroll
        for (int kc = 0; kc < 4; ++kc)
            aw[kc] = *(const v16h*)(wts + (size_t)(184 + kc) * 512 + (size_t)lane * 16);

        v8f acc = {};
#pragma unroll
        for (int kc = 0; kc < 4; ++kc) {
            const _Float16* p = &act[buf][mt * 16 + (lane & 15)][kc * 32 + koff2];
            union { v16h v; v8h h[2]; } bf;
            bf.h[0] = *(const v8h*)p;
            bf.h[1] = *(const v8h*)(p + 8);
            acc = __builtin_amdgcn_wmma_f32_16x16x32_f16(
                false, aw[kc], false, bf.v, (short)0, acc, false, false);
        }
        // D rows (M) = out features n = v + mbase; only lanes 0..15 (mbase=0) hold n<3.
        if (lane < 16) {
            const int m = rowBase + mt * 16 + lane;
            out[(size_t)m * 3 + 0] = acc[0] + b6[0];
            out[(size_t)m * 3 + 1] = acc[1] + b6[1];
            out[(size_t)m * 3 + 2] = acc[2] + b6[2];
        }
    }
}

// ---------------------------------------------------------------------------
extern "C" void kernel_launch(void* const* d_in, const int* in_sizes, int n_in,
                              void* d_out, int out_size, void* d_ws, size_t ws_size,
                              hipStream_t stream) {
    const float* xyz  = (const float*)d_in[0];
    const float* cond = (const float*)d_in[1];
    const float* emb  = (const float*)d_in[2];
    const float* W[7];
    const float* B[7];
    for (int i = 0; i < 7; ++i) {
        W[i] = (const float*)d_in[3 + 2 * i];
        B[i] = (const float*)d_in[4 + 2 * i];
    }
    _Float16* wts = (_Float16*)d_ws;   // needs 188 * 1024 bytes = 192.5 KB

    prep_weights<<<NBLOBS, 32, 0, stream>>>(W[0], W[1], W[2], W[3], W[4], W[5], W[6], wts);

    fused_encode_mlp<<<N_PTS / MTILE, 256, 0, stream>>>(
        xyz, cond, emb, wts,
        B[0], B[1], B[2], B[3], B[4], B[5], B[6],
        (float*)d_out);
}
